// SwinContextBlock_75823352644184
// MI455X (gfx1250) — matrix operand order repaired and checked by reading, not compiled, not run on hardware
//
#include <hip/hip_runtime.h>
#include <hip/hip_fp16.h>
#include <math.h>

// ---------------------------------------------------------------------------
// SwinContextBlock for MI455X (gfx1250): all GEMMs via v_wmma_f32_16x16x32_f16
// f16 inputs / f32 accumulate, 32-row M-tiles (halved L2 weight traffic),
// k-outer/t-inner WMMA bursts, fused epilogues.
// ---------------------------------------------------------------------------

typedef __attribute__((ext_vector_type(16))) _Float16 v16h;
typedef __attribute__((ext_vector_type(8)))  _Float16 half8;
typedef __attribute__((ext_vector_type(8)))  float    v8f;

static constexpr int BATCH = 16;
static constexpr int HWP   = 56 * 56;              // 3136 pixels per image
static constexpr int MROWS = BATCH * HWP;          // 50176
static constexpr int CIN   = 256;
static constexpr int DDIM  = 512;
static constexpr int MLPD  = 2048;
static constexpr int QKVD  = 3 * DDIM;             // 1536

union U16 { v16h v; half8 h[2]; };

// ---------------------------------------------------------------------------
// WMMA GEMM:  C[M,N] = A[M,K] * Wt[N,K]^T  (+ fused epilogue)
//   EPI 0: +bias, store f16                       (qkv)
//   EPI 1: +bias, SiLU, store f32 [M,N]           (in_proj conv+BN+SiLU)
//   EPI 2: +bias, accumulate into f32 out (+=)    (proj / mlp2 residual)
//   EPI 3: +bias, exact GELU, store f16           (mlp1)
//   EPI 4: +bias, SiLU, scatter to NCHW f32       (out_proj)
// grid.x = M/32 (32-row M-tile), grid.y = N / (NT*128) column split.
// block = 256 threads (8 waves). Each wave: 2 row-tiles x NT col-tiles.
// Per k-step: A fragments loaded once from LDS, each B fragment drives 2 WMMAs.
// ---------------------------------------------------------------------------
template<int EPI, int NT>
__global__ __launch_bounds__(256)
void gemm_wmma(const _Float16* __restrict__ A, const _Float16* __restrict__ Wt,
               const float* __restrict__ bias, void* __restrict__ outp,
               int N, int K)
{
    __shared__ _Float16 sA[32 * 512];          // 32-row A tile, K-chunk of 512
    const int m0   = blockIdx.x * 32;
    const int cb   = blockIdx.y * (NT * 128);  // column base for this block
    const int tid  = threadIdx.x;
    const int wave = tid >> 5;
    const int lane = tid & 31;
    const int sel  = lane >> 4;                // 0: lanes 0-15, 1: lanes 16-31
    const int rowA = lane & 15;
    const int colb = lane & 15;

    v8f acc[2][NT] = {};

    for (int kc = 0; kc < K; kc += 512) {
        int chunk = K - kc; if (chunk > 512) chunk = 512;
        __syncthreads();
        // cooperative stage of A tile (32 x chunk halves), 16B vectors
        const int vpr = chunk >> 3;            // half8 vectors per row
        for (int idx = tid; idx < (vpr << 5); idx += 256) {
            int r = idx / vpr;
            int c = idx - r * vpr;
            *(uint4*)(sA + r * 512 + c * 8) =
                *(const uint4*)(A + (size_t)(m0 + r) * K + kc + c * 8);
        }
        __syncthreads();

        for (int k0 = 0; k0 < chunk; k0 += 32) {
            // A fragments for both row-tiles (ISA 16-bit A 16x32 layout)
            U16 a0, a1;
            a0.h[0] = *(const half8*)(sA + rowA * 512 + k0 + sel * 8);
            a0.h[1] = *(const half8*)(sA + rowA * 512 + k0 + 16 + sel * 8);
            a1.h[0] = *(const half8*)(sA + (rowA + 16) * 512 + k0 + sel * 8);
            a1.h[1] = *(const half8*)(sA + (rowA + 16) * 512 + k0 + 16 + sel * 8);
#pragma unroll
            for (int t = 0; t < NT; ++t) {
                const int col = cb + t * 128 + wave * 16 + colb;
                const _Float16* wrow = Wt + (size_t)col * K + kc + k0;
                U16 b;
                b.h[0] = *(const half8*)(wrow + sel * 8);
                b.h[1] = *(const half8*)(wrow + 16 + sel * 8);
                if (k0 == 0 && kc + 512 < K)   // stream next K-chunk of W into L2/L0
                    __builtin_prefetch(wrow + 512, 0, 1);
                acc[0][t] = __builtin_amdgcn_wmma_f32_16x16x32_f16(
                    false, a0.v, false, b.v, (short)0, acc[0][t], false, false);
                acc[1][t] = __builtin_amdgcn_wmma_f32_16x16x32_f16(
                    false, a1.v, false, b.v, (short)0, acc[1][t], false, false);
            }
        }
    }

    // epilogue: row-tile rt rows = m0 + rt*16 + r + 8*sel (C/D layout)
#pragma unroll
    for (int t = 0; t < NT; ++t) {
        const int col = cb + t * 128 + wave * 16 + colb;
        const float bv = bias[col];
#pragma unroll
        for (int rt = 0; rt < 2; ++rt) {
#pragma unroll
            for (int r = 0; r < 8; ++r) {
                const int m = m0 + rt * 16 + r + 8 * sel;
                float v = acc[rt][t][r] + bv;
                if (EPI == 0) {
                    ((_Float16*)outp)[(size_t)m * N + col] = (_Float16)v;
                } else if (EPI == 1) {
                    ((float*)outp)[(size_t)m * N + col] = v / (1.f + __expf(-v));
                } else if (EPI == 2) {
                    ((float*)outp)[(size_t)m * N + col] += v;
                } else if (EPI == 3) {
                    float g = 0.5f * v * (1.f + erff(v * 0.70710678118f));
                    ((_Float16*)outp)[(size_t)m * N + col] = (_Float16)g;
                } else { // EPI == 4: scatter [M,N] -> NCHW
                    float s = v / (1.f + __expf(-v));
                    int b  = m / HWP;
                    int hw = m - b * HWP;
                    ((float*)outp)[((size_t)b * N + col) * HWP + hw] = s;
                }
            }
        }
    }
}

// ---------------------------------------------------------------------------
// LayerNorm over D=512, one row per 128-thread block; writes f16.
// ---------------------------------------------------------------------------
__global__ __launch_bounds__(128)
void ln512_kernel(const float* __restrict__ x, const float* __restrict__ g,
                  const float* __restrict__ b, _Float16* __restrict__ out)
{
    __shared__ float red[8];
    const size_t base = (size_t)blockIdx.x * 512;
    const int tid = threadIdx.x;
    float v0 = x[base + tid], v1 = x[base + tid + 128];
    float v2 = x[base + tid + 256], v3 = x[base + tid + 384];
    float s = v0 + v1 + v2 + v3;
    for (int o = 16; o; o >>= 1) s += __shfl_xor(s, o, 32);
    if ((tid & 31) == 0) red[tid >> 5] = s;
    __syncthreads();
    const float mean = (red[0] + red[1] + red[2] + red[3]) * (1.f / 512.f);
    float d0 = v0 - mean, d1 = v1 - mean, d2 = v2 - mean, d3 = v3 - mean;
    float s2 = d0 * d0 + d1 * d1 + d2 * d2 + d3 * d3;
    for (int o = 16; o; o >>= 1) s2 += __shfl_xor(s2, o, 32);
    if ((tid & 31) == 0) red[4 + (tid >> 5)] = s2;
    __syncthreads();
    const float var = (red[4] + red[5] + red[6] + red[7]) * (1.f / 512.f);
    const float inv = rsqrtf(var + 1e-5f);
    out[base + tid]       = (_Float16)(d0 * inv * g[tid]       + b[tid]);
    out[base + tid + 128] = (_Float16)(d1 * inv * g[tid + 128] + b[tid + 128]);
    out[base + tid + 256] = (_Float16)(d2 * inv * g[tid + 256] + b[tid + 256]);
    out[base + tid + 384] = (_Float16)(d3 * inv * g[tid + 384] + b[tid + 384]);
}

// ---------------------------------------------------------------------------
// Shifted-window attention. grid = (1024 windows, 8 heads), block = 256.
// qkv f16 [M, 1536] in natural (b,h,w) order; roll folded into index math.
// ---------------------------------------------------------------------------
__device__ __forceinline__ int swreg(int c) { return c < 49 ? 0 : (c < 53 ? 1 : 2); }

__global__ __launch_bounds__(256)
void attn_kernel(const _Float16* __restrict__ qkv, const float* __restrict__ rel,
                 _Float16* __restrict__ outp, int shift)
{
    __shared__ float sq[49 * 65], sk[49 * 65], sv[49 * 65]; // pad 65: no bank conflict
    __shared__ float sp[8][52];
    const int win = blockIdx.x, head = blockIdx.y;
    const int b = win >> 6, w = win & 63, wh = w >> 3, ww = w & 7;
    const int tid = threadIdx.x;

    for (int idx = tid; idx < 49 * 64; idx += 256) {
        int j = idx >> 6, d = idx & 63;
        int ih = j / 7, iw = j - ih * 7;
        int sh = wh * 7 + ih + shift; if (sh >= 56) sh -= 56;
        int sw = ww * 7 + iw + shift; if (sw >= 56) sw -= 56;
        size_t row = ((size_t)(b * 56 + sh) * 56 + sw) * (size_t)QKVD;
        int cc = head * 64 + d;
        sq[j * 65 + d] = (float)qkv[row + cc] * 0.125f;          // q * hd^-0.5
        sk[j * 65 + d] = (float)qkv[row + 512 + cc];
        sv[j * 65 + d] = (float)qkv[row + 1024 + cc];
    }
    __syncthreads();

    const int wave = tid >> 5, lane = tid & 31;
    for (int i = wave; i < 49; i += 8) {
        const int yi = i / 7, xi = i - yi * 7;
        const int li = 3 * swreg(wh * 7 + yi) + swreg(ww * 7 + xi);
        const int j0 = lane, j1 = lane + 32;
        float s0, s1 = -1e30f;
        {
            float a = 0.f;
            for (int d = 0; d < 64; ++d) a += sq[i * 65 + d] * sk[j0 * 65 + d];
            int yj = j0 / 7, xj = j0 - yj * 7;
            s0 = a + rel[((yi - yj + 6) * 13 + (xi - xj + 6)) * 8 + head];
            if (shift) {
                int lj = 3 * swreg(wh * 7 + yj) + swreg(ww * 7 + xj);
                if (li != lj) s0 -= 100.f;
            }
        }
        if (j1 < 49) {
            float a = 0.f;
            for (int d = 0; d < 64; ++d) a += sq[i * 65 + d] * sk[j1 * 65 + d];
            int yj = j1 / 7, xj = j1 - yj * 7;
            s1 = a + rel[((yi - yj + 6) * 13 + (xi - xj + 6)) * 8 + head];
            if (shift) {
                int lj = 3 * swreg(wh * 7 + yj) + swreg(ww * 7 + xj);
                if (li != lj) s1 -= 100.f;
            }
        }
        float mx = fmaxf(s0, s1);
        for (int o = 16; o; o >>= 1) mx = fmaxf(mx, __shfl_xor(mx, o, 32));
        float e0 = __expf(s0 - mx);
        float e1 = (j1 < 49) ? __expf(s1 - mx) : 0.f;
        float sum = e0 + e1;
        for (int o = 16; o; o >>= 1) sum += __shfl_xor(sum, o, 32);
        float inv = 1.f / sum;
        sp[wave][j0] = e0 * inv;
        if (j1 < 49) sp[wave][j1] = e1 * inv;

        // out[i, d] = sum_j p[j] * v[j, d]; lanes own d = lane, lane+32
        float o0 = 0.f, o1 = 0.f;
        for (int j = 0; j < 49; ++j) {
            float p = sp[wave][j];
            o0 += p * sv[j * 65 + lane];
            o1 += p * sv[j * 65 + lane + 32];
        }
        int sh2 = wh * 7 + yi + shift; if (sh2 >= 56) sh2 -= 56;
        int sw2 = ww * 7 + xi + shift; if (sw2 >= 56) sw2 -= 56;
        size_t orow = ((size_t)(b * 56 + sh2) * 56 + sw2) * (size_t)DDIM + head * 64;
        outp[orow + lane]      = (_Float16)o0;
        outp[orow + lane + 32] = (_Float16)o1;
    }
}

// ---------------------------------------------------------------------------
// Prep kernels
// ---------------------------------------------------------------------------
__global__ void pack_x_kernel(const float* __restrict__ x, _Float16* __restrict__ a,
                              int total) // NCHW f32 -> [B*HW, C] f16
{
    int idx = blockIdx.x * 256 + threadIdx.x;
    if (idx >= total) return;
    int b   = idx / (CIN * HWP);
    int rem = idx - b * (CIN * HWP);
    int c   = rem / HWP;
    int hw  = rem - c * HWP;
    a[((size_t)b * HWP + hw) * CIN + c] = (_Float16)x[idx];
}

__global__ void fold_conv_kernel(const float* __restrict__ w, const float* __restrict__ gamma,
                                 const float* __restrict__ beta, const float* __restrict__ mean,
                                 const float* __restrict__ var,
                                 _Float16* __restrict__ w16, float* __restrict__ biasf,
                                 int O, int CI)
{
    int idx = blockIdx.x * 256 + threadIdx.x;
    if (idx >= O * CI) return;
    int o = idx / CI;
    float inv = rsqrtf(var[o] + 1e-5f) * gamma[o];
    w16[idx] = (_Float16)(w[idx] * inv);
    if (idx - o * CI == 0) biasf[o] = beta[o] - mean[o] * inv;
}

__global__ void cvt_f16_kernel(const float* __restrict__ in, _Float16* __restrict__ out, int n)
{
    int idx = blockIdx.x * 256 + threadIdx.x;
    if (idx < n) out[idx] = (_Float16)in[idx];
}

// ---------------------------------------------------------------------------
// Orchestration
// ---------------------------------------------------------------------------
extern "C" void kernel_launch(void* const* d_in, const int* in_sizes, int n_in,
                              void* d_out, int out_size, void* d_ws, size_t ws_size,
                              hipStream_t stream)
{
    (void)in_sizes; (void)n_in; (void)out_size; (void)ws_size;
    // d_in order (setup_inputs insertion order):
    // 0:x  1-5: in_proj {w,gamma,beta,mean,var}
    // 6+13k: block k {ln1_g,ln1_b,qkv_w,qkv_b,rel_table,proj_w,proj_b,
    //                 ln2_g,ln2_b,mlp_w1,mlp_b1,mlp_w2,mlp_b2}
    // 32-36: out_proj {w,gamma,beta,mean,var}
    const float* x     = (const float*)d_in[0];
    const float* in_w  = (const float*)d_in[1];
    const float* in_g  = (const float*)d_in[2];
    const float* in_b  = (const float*)d_in[3];
    const float* in_m  = (const float*)d_in[4];
    const float* in_v  = (const float*)d_in[5];
    const float* out_w = (const float*)d_in[32];
    const float* out_g = (const float*)d_in[33];
    const float* out_b = (const float*)d_in[34];
    const float* out_m = (const float*)d_in[35];
    const float* out_v = (const float*)d_in[36];

    // workspace carve-up
    char* p = (char*)d_ws;
    auto alloc = [&](size_t bytes) { void* r = (void*)p; p += (bytes + 255) & ~(size_t)255; return r; };
    float*    h       = (float*)   alloc((size_t)MROWS * DDIM * 4);   // residual stream (f32)
    _Float16* bufA    = (_Float16*)alloc((size_t)MROWS * MLPD * 2);   // xn1 / attn_out / mlp_mid / h16
    _Float16* bufB    = (_Float16*)alloc((size_t)MROWS * QKVD * 2);   // packed x / qkv / xn2
    _Float16* w_in16  = (_Float16*)alloc((size_t)DDIM * CIN * 2);
    _Float16* w_out16 = (_Float16*)alloc((size_t)CIN * DDIM * 2);
    float*    bias_in  = (float*)  alloc(DDIM * 4);
    float*    bias_out = (float*)  alloc(CIN * 4);
    _Float16* wq16[2]; _Float16* wp16[2]; _Float16* wm116[2]; _Float16* wm216[2];
    for (int k = 0; k < 2; ++k) {
        wq16[k]  = (_Float16*)alloc((size_t)QKVD * DDIM * 2);
        wp16[k]  = (_Float16*)alloc((size_t)DDIM * DDIM * 2);
        wm116[k] = (_Float16*)alloc((size_t)MLPD * DDIM * 2);
        wm216[k] = (_Float16*)alloc((size_t)DDIM * MLPD * 2);
    }

    auto grid1 = [](int n) { return (n + 255) / 256; };

    // ---- weight prep (f16 convert; BN folded into conv weights/bias) ----
    fold_conv_kernel<<<grid1(DDIM * CIN), 256, 0, stream>>>(in_w, in_g, in_b, in_m, in_v,
                                                            w_in16, bias_in, DDIM, CIN);
    fold_conv_kernel<<<grid1(CIN * DDIM), 256, 0, stream>>>(out_w, out_g, out_b, out_m, out_v,
                                                            w_out16, bias_out, CIN, DDIM);
    for (int k = 0; k < 2; ++k) {
        const int base = 6 + 13 * k;
        cvt_f16_kernel<<<grid1(QKVD * DDIM), 256, 0, stream>>>((const float*)d_in[base + 2],  wq16[k],  QKVD * DDIM);
        cvt_f16_kernel<<<grid1(DDIM * DDIM), 256, 0, stream>>>((const float*)d_in[base + 5],  wp16[k],  DDIM * DDIM);
        cvt_f16_kernel<<<grid1(MLPD * DDIM), 256, 0, stream>>>((const float*)d_in[base + 9],  wm116[k], MLPD * DDIM);
        cvt_f16_kernel<<<grid1(DDIM * MLPD), 256, 0, stream>>>((const float*)d_in[base + 11], wm216[k], DDIM * MLPD);
    }

    const int GB = MROWS / 32; // 1568 GEMM blocks (32-row M-tiles)

    // ---- in_proj: pack NCHW -> [M,256] f16, then conv+BN+SiLU -> h (f32) ----
    pack_x_kernel<<<grid1(BATCH * CIN * HWP), 256, 0, stream>>>(x, bufB, BATCH * CIN * HWP);
    gemm_wmma<1, 4><<<dim3(GB, 1), 256, 0, stream>>>(bufB, w_in16, bias_in, h, DDIM, CIN);

    // ---- two Swin blocks ----
    for (int k = 0; k < 2; ++k) {
        const int base  = 6 + 13 * k;
        const int shift = (k == 0) ? 0 : 3;
        const float* ln1_g = (const float*)d_in[base + 0];
        const float* ln1_b = (const float*)d_in[base + 1];
        const float* qkv_b = (const float*)d_in[base + 3];
        const float* rel   = (const float*)d_in[base + 4];
        const float* prj_b = (const float*)d_in[base + 6];
        const float* ln2_g = (const float*)d_in[base + 7];
        const float* ln2_b = (const float*)d_in[base + 8];
        const float* m1_b  = (const float*)d_in[base + 10];
        const float* m2_b  = (const float*)d_in[base + 12];

        ln512_kernel<<<MROWS, 128, 0, stream>>>(h, ln1_g, ln1_b, bufA);                    // xn1 f16
        gemm_wmma<0, 6><<<dim3(GB, 2), 256, 0, stream>>>(bufA, wq16[k], qkv_b, bufB, QKVD, DDIM);
        attn_kernel<<<dim3(1024, 8, 1), 256, 0, stream>>>(bufB, rel, bufA, shift);         // attn_out f16
        gemm_wmma<2, 4><<<dim3(GB, 1), 256, 0, stream>>>(bufA, wp16[k], prj_b, h, DDIM, DDIM);
        ln512_kernel<<<MROWS, 128, 0, stream>>>(h, ln2_g, ln2_b, bufB);                    // xn2 f16
        gemm_wmma<3, 8><<<dim3(GB, 2), 256, 0, stream>>>(bufB, wm116[k], m1_b, bufA, MLPD, DDIM);
        gemm_wmma<2, 4><<<dim3(GB, 1), 256, 0, stream>>>(bufA, wm216[k], m2_b, h, DDIM, MLPD);
    }

    // ---- out_proj: h -> f16, conv+BN+SiLU scatter to NCHW f32 output ----
    cvt_f16_kernel<<<grid1(MROWS * DDIM), 256, 0, stream>>>(h, bufA, MROWS * DDIM);
    gemm_wmma<4, 2><<<dim3(GB, 1), 256, 0, stream>>>(bufA, w_out16, bias_out, d_out, CIN, DDIM);
}
